// VRAE_67216238182740
// MI455X (gfx1250) — matrix operand order repaired and checked by reading, not compiled, not run on hardware
//
#include <hip/hip_runtime.h>
#include <hip/hip_fp16.h>
#include <cstdint>
#include <cstddef>

// ---------------------------------------------------------------------------
// VRAE forward for MI455X (gfx1250, wave32).
// - All GEMMs: v_wmma_f32_16x16x32_f16 (f16 in, f32 accumulate).
// - "Gate-aligned" N-tile mapping: wave w owns N-tiles {w, w+8, w+16}, i.e.
//   the SAME column block of the r, z, n gates -> GRU gate math is done
//   entirely in accumulator registers (no LDS gate staging at all).
// - Encoder layers 1-2: input projection gi precomputed by a parallel WMMA
//   GEMM (M=64 tiles for 4x B-fragment reuse); scans do only the recurrent
//   GEMM with register-resident Whh fragments.
// ---------------------------------------------------------------------------

#define DEV __device__ __forceinline__

typedef __attribute__((ext_vector_type(16))) _Float16 v16h;
typedef __attribute__((ext_vector_type(8)))  _Float16 v8h;
typedef __attribute__((ext_vector_type(8)))  float    v8f;

namespace vrae {
constexpr int B  = 256;
constexpr int T  = 512;
constexpr int D  = 32;
constexpr int H  = 128;
constexpr int L  = 32;
constexpr int G3 = 3 * H;   // 384 gate columns (r,z,n)
}

// --- WMMA fragment loaders (layouts per CDNA5 ISA §7.12.2) -------
DEV v16h load_a_frag(const _Float16* row, int kbase, int half) {
  const _Float16* p0 = row + kbase + half * 8;
  const _Float16* p1 = row + kbase + 16 + half * 8;
  v8h lo = *(const v8h*)p0;
  v8h hi = *(const v8h*)p1;
  v16h a;
#pragma unroll
  for (int j = 0; j < 8; ++j) { a[j] = lo[j]; a[8 + j] = hi[j]; }
  return a;
}

DEV v16h load_b_frag(const _Float16* W, int K, int ncol, int kbase, int half) {
  const _Float16* p = W + (size_t)ncol * K + kbase + half * 8;
  v8h lo = *(const v8h*)p;
  v8h hi = *(const v8h*)(p + 16);
  v16h b;
#pragma unroll
  for (int j = 0; j < 8; ++j) { b[j] = lo[j]; b[8 + j] = hi[j]; }
  return b;
}

DEV v8f wmma_f16(v16h a, v16h b, v8f c) {
  return __builtin_amdgcn_wmma_f32_16x16x32_f16(false, a, false, b, (short)0, c,
                                                false, false);
}

DEV float sigmoidf_(float x) { return 1.f / (1.f + __expf(-x)); }

// Gate-aligned column for gate q of this wave: cols q*128 + wave*16 + mA.
DEV int gate_ncol(int wave, int q, int mA) { return (q * 8 + wave) * 16 + mA; }

// Preload this wave's gate-aligned B fragments (loop-invariant weights).
template <int NKT>
DEV void preload_b3g(const _Float16* W, int K, int wave, int mA, int half,
                     v16h* f) {
#pragma unroll
  for (int q = 0; q < 3; ++q)
#pragma unroll
    for (int kt = 0; kt < NKT; ++kt)
      f[q * NKT + kt] = load_b_frag(W, K, gate_ncol(wave, q, mA), kt * 32, half);
}

// Accumulate A @ W^T into acc[0..2] (gate-aligned tiles), B streamed.
DEV void gemm3_acc_stream(const _Float16* Arow, int K, const _Float16* W,
                          int wave, int mA, int half, v8f acc[3]) {
  for (int kt = 0; kt < K; kt += 32) {
    v16h a = load_a_frag(Arow, kt, half);
#pragma unroll
    for (int q = 0; q < 3; ++q)
      acc[q] = wmma_f16(a, load_b_frag(W, K, gate_ncol(wave, q, mA), kt, half),
                        acc[q]);
  }
}

// --------------------------- utility kernels --------------------------------
__global__ void f32_to_f16_kernel(const float* __restrict__ s,
                                  _Float16* __restrict__ d, size_t n) {
  size_t i = (size_t)blockIdx.x * blockDim.x + threadIdx.x;
  size_t stride = (size_t)gridDim.x * blockDim.x;
  for (; i < n; i += stride) d[i] = (_Float16)s[i];
}

__global__ void zero_u32_kernel(uint32_t* __restrict__ d, size_t n32) {
  size_t i = (size_t)blockIdx.x * blockDim.x + threadIdx.x;
  size_t stride = (size_t)gridDim.x * blockDim.x;
  for (; i < n32; i += stride) d[i] = 0u;
}

// ----------------- parallel input-projection GEMM (layers 1-2) --------------
// gi[dir, r, :] = seq[r, :] @ Wih[dir]^T, M=64 tile per workgroup so each
// B fragment is reused by 4 M-subtiles (L2-bandwidth relief).
__global__ __launch_bounds__(256) void gi_gemm_kernel(
    const _Float16* __restrict__ seq_in,   // [B*T, 2H]
    const _Float16* __restrict__ Wih,      // [2, 384, 2H]
    _Float16* __restrict__ gi) {           // [2, B*T, 384]
  using namespace vrae;
  const int mt = blockIdx.x, dir = blockIdx.y;    // 64-row tile index
  const int tid = threadIdx.x;
  const int lane = tid & 31, wave = tid >> 5;
  const int half = lane >> 4, mA = lane & 15;
  const int K = 2 * H;

  const _Float16* Wi = Wih + (size_t)dir * G3 * K;
  const size_t row0 = (size_t)mt * 64 + mA;

  v8f acc[4][3] = {};
#pragma unroll
  for (int kt = 0; kt < 2 * H; kt += 32) {
    v16h bf[3];
#pragma unroll
    for (int q = 0; q < 3; ++q)
      bf[q] = load_b_frag(Wi, K, (wave * 3 + q) * 16 + mA, kt, half);
#pragma unroll
    for (int s = 0; s < 4; ++s) {
      v16h a = load_a_frag(seq_in + (row0 + s * 16) * K, kt, half);
#pragma unroll
      for (int q = 0; q < 3; ++q) acc[s][q] = wmma_f16(a, bf[q], acc[s][q]);
    }
  }
#pragma unroll
  for (int s = 0; s < 4; ++s) {
    _Float16* grow = gi + ((size_t)dir * B * T + (size_t)mt * 64 + s * 16) * G3;
#pragma unroll
    for (int q = 0; q < 3; ++q) {
      int n = (wave * 3 + q) * 16 + mA;
#pragma unroll
      for (int v = 0; v < 8; ++v)
        grow[(size_t)(half * 8 + v) * G3 + n] = (_Float16)acc[s][q][v];
    }
  }
}

// -------------------- encoder layer-0 scan (fused gi, K=32) -----------------
// grid = (B/16, 2 dirs). Wih/Whh fragments register-resident; gate math in
// accumulator registers; only h lives in LDS (A source for the next step).
__global__ __launch_bounds__(256) void gru_scan_l0_kernel(
    const _Float16* __restrict__ x16,      // [B,T,32]
    _Float16* __restrict__ seq_out,        // [B,T,2H]
    const _Float16* __restrict__ Wih,      // [2,384,32]
    const _Float16* __restrict__ Whh,      // [2,384,128]
    const float* __restrict__ bih, const float* __restrict__ bhh,
    const int* __restrict__ lengths) {
  using namespace vrae;
  __shared__ __align__(16) _Float16 h_s[16 * H];
  __shared__ int len_s[16];

  const int tile = blockIdx.x, dir = blockIdx.y;
  const int tid = threadIdx.x;
  const int lane = tid & 31, wave = tid >> 5;
  const int half = lane >> 4, mA = lane & 15;
  const int rowbase = tile * 16;
  const int j = wave * 16 + mA;            // this lane's hidden column

  v16h wih_f[3], whh_f[12];
  preload_b3g<1>(Wih + (size_t)dir * G3 * D, D, wave, mA, half, wih_f);
  preload_b3g<4>(Whh + (size_t)dir * G3 * H, H, wave, mA, half, whh_f);

  const float* bi = bih + dir * G3;
  const float* bh = bhh + dir * G3;
  const float br = bi[j] + bh[j];
  const float bz = bi[H + j] + bh[H + j];
  const float bin_ = bi[2 * H + j];
  const float bhn = bh[2 * H + j];

  if (tid < 16) len_s[tid] = lengths[rowbase + tid];
  for (int i = tid; i < 16 * H; i += 256) h_s[i] = (_Float16)0.f;
  __syncthreads();

  int lenv[8];
  float hprev[8];
#pragma unroll
  for (int v = 0; v < 8; ++v) { lenv[v] = len_s[half * 8 + v]; hprev[v] = 0.f; }
  const int lenA = len_s[mA];
  const _Float16* hrow = &h_s[(size_t)mA * H];

  for (int t = 0; t < T; ++t) {
    const int sA = dir ? ((t < lenA) ? (lenA - 1 - t) : 0) : t;
    const _Float16* xrow = x16 + ((size_t)(rowbase + mA) * T + sA) * D;

    // acc[q] = h @ Whh^T (gate q); then fold x @ Wih^T into r,z; n kept apart
    v8f acc[3] = {};
#pragma unroll
    for (int kt = 0; kt < 4; ++kt) {
      v16h a = load_a_frag(hrow, kt * 32, half);
#pragma unroll
      for (int q = 0; q < 3; ++q) acc[q] = wmma_f16(a, whh_f[q * 4 + kt], acc[q]);
    }
    v16h ax = load_a_frag(xrow, 0, half);
    acc[0] = wmma_f16(ax, wih_f[0], acc[0]);
    acc[1] = wmma_f16(ax, wih_f[1], acc[1]);
    v8f accn = {};
    accn = wmma_f16(ax, wih_f[2], accn);
    __syncthreads();   // all h_s reads complete before updates

#pragma unroll
    for (int v = 0; v < 8; ++v) {
      const int m = half * 8 + v;
      const bool act = (t < lenv[v]);
      float r = sigmoidf_(acc[0][v] + br);
      float z = sigmoidf_(acc[1][v] + bz);
      float n = tanhf(accn[v] + bin_ + r * (acc[2][v] + bhn));
      float hnew = (1.f - z) * n + z * hprev[v];
      float hk = act ? hnew : hprev[v];
      hprev[v] = hk;
      h_s[m * H + j] = (_Float16)hk;
      if (dir == 0)
        seq_out[(((size_t)rowbase + m) * T + t) * (2 * H) + j] =
            act ? (_Float16)hnew : (_Float16)0.f;
      else if (act)
        seq_out[(((size_t)rowbase + m) * T + (size_t)(lenv[v] - 1 - t)) * (2 * H) +
                H + j] = (_Float16)hnew;
    }
    __syncthreads();
  }
}

// --------------- encoder layers 1-2 scan (gi precomputed) -------------------
__global__ __launch_bounds__(256) void gru_scan_pre_kernel(
    const _Float16* __restrict__ gi,       // [2, B*T, 384]
    _Float16* __restrict__ seq_out,        // nullable, [B,T,2H]
    const _Float16* __restrict__ Whh,      // [2,384,128]
    const float* __restrict__ bih, const float* __restrict__ bhh,
    const int* __restrict__ lengths,
    float* __restrict__ hfinal) {          // nullable, [2,B,128]
  using namespace vrae;
  __shared__ __align__(16) _Float16 h_s[16 * H];
  __shared__ int len_s[16];

  const int tile = blockIdx.x, dir = blockIdx.y;
  const int tid = threadIdx.x;
  const int lane = tid & 31, wave = tid >> 5;
  const int half = lane >> 4, mA = lane & 15;
  const int rowbase = tile * 16;
  const int j = wave * 16 + mA;

  v16h whh_f[12];
  preload_b3g<4>(Whh + (size_t)dir * G3 * H, H, wave, mA, half, whh_f);

  const float* bi = bih + dir * G3;
  const float* bh = bhh + dir * G3;
  const float br = bi[j] + bh[j];
  const float bz = bi[H + j] + bh[H + j];
  const float bin_ = bi[2 * H + j];
  const float bhn = bh[2 * H + j];

  if (tid < 16) len_s[tid] = lengths[rowbase + tid];
  for (int i = tid; i < 16 * H; i += 256) h_s[i] = (_Float16)0.f;
  __syncthreads();

  int lenv[8];
  float hprev[8];
#pragma unroll
  for (int v = 0; v < 8; ++v) { lenv[v] = len_s[half * 8 + v]; hprev[v] = 0.f; }
  const _Float16* hrow = &h_s[(size_t)mA * H];
  const _Float16* giB = gi + (size_t)dir * B * T * G3;

  for (int t = 0; t < T; ++t) {
    // prefetch this lane's precomputed input gates (8 rows x 3 gates)
    _Float16 gir[8], giz[8], gin[8];
#pragma unroll
    for (int v = 0; v < 8; ++v) {
      const int m = half * 8 + v;
      const int lv = lenv[v];
      const int s = dir ? ((t < lv) ? (lv - 1 - t) : 0) : t;
      const _Float16* row = giB + ((size_t)(rowbase + m) * T + s) * G3;
      gir[v] = row[j];
      giz[v] = row[H + j];
      gin[v] = row[2 * H + j];
    }

    v8f acc[3] = {};
#pragma unroll
    for (int kt = 0; kt < 4; ++kt) {
      v16h a = load_a_frag(hrow, kt * 32, half);
#pragma unroll
      for (int q = 0; q < 3; ++q) acc[q] = wmma_f16(a, whh_f[q * 4 + kt], acc[q]);
    }
    __syncthreads();

#pragma unroll
    for (int v = 0; v < 8; ++v) {
      const int m = half * 8 + v;
      const bool act = (t < lenv[v]);
      float r = sigmoidf_((float)gir[v] + acc[0][v] + br);
      float z = sigmoidf_((float)giz[v] + acc[1][v] + bz);
      float n = tanhf((float)gin[v] + bin_ + r * (acc[2][v] + bhn));
      float hnew = (1.f - z) * n + z * hprev[v];
      float hk = act ? hnew : hprev[v];
      hprev[v] = hk;
      h_s[m * H + j] = (_Float16)hk;
      if (seq_out) {
        if (dir == 0)
          seq_out[(((size_t)rowbase + m) * T + t) * (2 * H) + j] =
              act ? (_Float16)hnew : (_Float16)0.f;
        else if (act)
          seq_out[(((size_t)rowbase + m) * T + (size_t)(lenv[v] - 1 - t)) *
                      (2 * H) + H + j] = (_Float16)hnew;
      }
    }
    __syncthreads();
  }

  if (hfinal) {
#pragma unroll
    for (int v = 0; v < 8; ++v)
      hfinal[((size_t)dir * B + rowbase + half * 8 + v) * H + j] = hprev[v];
  }
}

// --------------------------- latent head kernel -----------------------------
__global__ __launch_bounds__(128) void latent_kernel(
    const float* __restrict__ hfinal, const float* __restrict__ eps,
    const float* __restrict__ Wmean, const float* __restrict__ bmean,
    const float* __restrict__ Wlogvar, const float* __restrict__ blogvar,
    const float* __restrict__ Wz2h, const float* __restrict__ bz2h,
    float* __restrict__ mean_out, float* __restrict__ logvar_out,
    float* __restrict__ hinit) {
  using namespace vrae;
  __shared__ float hcat_s[2 * H];
  __shared__ float mean_s[L], logvar_s[L], z_s[L];
  const int b = blockIdx.x, tid = threadIdx.x;
  hcat_s[tid]     = hfinal[(size_t)b * H + tid];
  hcat_s[H + tid] = hfinal[((size_t)B + b) * H + tid];
  __syncthreads();
  if (tid < L) {
    float acc = bmean[tid];
    for (int k = 0; k < 2 * H; ++k) acc += hcat_s[k] * Wmean[tid * 2 * H + k];
    mean_s[tid] = acc;
    mean_out[(size_t)b * L + tid] = acc;
  } else if (tid < 2 * L) {
    int j = tid - L;
    float acc = blogvar[j];
    for (int k = 0; k < 2 * H; ++k) acc += hcat_s[k] * Wlogvar[j * 2 * H + k];
    logvar_s[j] = acc;
    logvar_out[(size_t)b * L + j] = acc;
  }
  __syncthreads();
  if (tid < L)
    z_s[tid] = mean_s[tid] + __expf(0.5f * logvar_s[tid]) * eps[(size_t)b * L + tid];
  __syncthreads();
  float acc = bz2h[tid];
  for (int k = 0; k < L; ++k) acc += z_s[k] * Wz2h[tid * L + k];
  hinit[(size_t)b * H + tid] = tanhf(acc);
}

// ------------------------ autoregressive decoder ----------------------------
// grid = B/16 x 256; gate math fully in registers; Whh0 fragments preloaded,
// stage-2 weights streamed (keeps VGPR pressure < 256).
__global__ __launch_bounds__(256) void decoder_kernel(
    const float* __restrict__ hinit,
    const _Float16* __restrict__ Wih0, const _Float16* __restrict__ Whh0,
    const _Float16* __restrict__ Wih1, const _Float16* __restrict__ Whh1,
    const _Float16* __restrict__ Wout,           // [D,H]
    const float* __restrict__ bih0, const float* __restrict__ bhh0,
    const float* __restrict__ bih1, const float* __restrict__ bhh1,
    const float* __restrict__ bout,
    float* __restrict__ recon) {                 // [B,T,D]
  using namespace vrae;
  __shared__ __align__(16) _Float16 h1_s[16 * H];
  __shared__ __align__(16) _Float16 h2_s[16 * H];
  __shared__ __align__(16) _Float16 din_s[16 * D];
  __shared__ float pred_s[16 * D];

  const int tile = blockIdx.x, tid = threadIdx.x;
  const int lane = tid & 31, wave = tid >> 5;
  const int half = lane >> 4, mA = lane & 15;
  const int rowbase = tile * 16;
  const int j = wave * 16 + mA;

  v16h whh0_f[12];
  preload_b3g<4>(Whh0, H, wave, mA, half, whh0_f);

  const float b0r = bih0[j] + bhh0[j];
  const float b0z = bih0[H + j] + bhh0[H + j];
  const float b0in = bih0[2 * H + j], b0hn = bhh0[2 * H + j];
  const float b1r = bih1[j] + bhh1[j];
  const float b1z = bih1[H + j] + bhh1[H + j];
  const float b1in = bih1[2 * H + j], b1hn = bhh1[2 * H + j];

  float h1p[8], h2p[8];
#pragma unroll
  for (int v = 0; v < 8; ++v) {
    float hv = hinit[((size_t)rowbase + half * 8 + v) * H + j];
    h1p[v] = hv;
    h2p[v] = hv;
  }
  for (int i = tid; i < 16 * H; i += 256) {
    _Float16 v = (_Float16)hinit[((size_t)rowbase + (i >> 7)) * H + (i & (H - 1))];
    h1_s[i] = v;
    h2_s[i] = v;
  }
  for (int i = tid; i < 16 * D; i += 256) din_s[i] = (_Float16)0.f;
  __syncthreads();

  const _Float16* dinrow = &din_s[(size_t)mA * D];
  const _Float16* h1row  = &h1_s[(size_t)mA * H];
  const _Float16* h2row  = &h2_s[(size_t)mA * H];

  for (int t = 0; t < T; ++t) {
    // ---- layer 1: gh via preloaded Whh0; gi (K=32) streamed, r/z folded ----
    v8f acc[3] = {};
#pragma unroll
    for (int kt = 0; kt < 4; ++kt) {
      v16h a = load_a_frag(h1row, kt * 32, half);
#pragma unroll
      for (int q = 0; q < 3; ++q) acc[q] = wmma_f16(a, whh0_f[q * 4 + kt], acc[q]);
    }
    v16h ax = load_a_frag(dinrow, 0, half);
    acc[0] = wmma_f16(ax, load_b_frag(Wih0, D, gate_ncol(wave, 0, mA), 0, half), acc[0]);
    acc[1] = wmma_f16(ax, load_b_frag(Wih0, D, gate_ncol(wave, 1, mA), 0, half), acc[1]);
    v8f accn = {};
    accn = wmma_f16(ax, load_b_frag(Wih0, D, gate_ncol(wave, 2, mA), 0, half), accn);
    __syncthreads();   // h1_s/din_s reads done
#pragma unroll
    for (int v = 0; v < 8; ++v) {
      float r = sigmoidf_(acc[0][v] + b0r);
      float z = sigmoidf_(acc[1][v] + b0z);
      float n = tanhf(accn[v] + b0in + r * (acc[2][v] + b0hn));
      float hnew = (1.f - z) * n + z * h1p[v];
      h1p[v] = hnew;
      h1_s[(half * 8 + v) * H + j] = (_Float16)hnew;
    }
    __syncthreads();   // h1 ready

    // ---- layer 2: gh (Whh1 streamed) + gi (Wih1 streamed), r/z folded ----
    v8f acc2[3] = {};
    gemm3_acc_stream(h2row, H, Whh1, wave, mA, half, acc2);
    v8f accn2 = {};
#pragma unroll
    for (int kt = 0; kt < H; kt += 32) {
      v16h a = load_a_frag(h1row, kt, half);
      acc2[0] = wmma_f16(a, load_b_frag(Wih1, H, gate_ncol(wave, 0, mA), kt, half), acc2[0]);
      acc2[1] = wmma_f16(a, load_b_frag(Wih1, H, gate_ncol(wave, 1, mA), kt, half), acc2[1]);
      accn2   = wmma_f16(a, load_b_frag(Wih1, H, gate_ncol(wave, 2, mA), kt, half), accn2);
    }
    __syncthreads();   // h1_s/h2_s reads done
#pragma unroll
    for (int v = 0; v < 8; ++v) {
      float r = sigmoidf_(acc2[0][v] + b1r);
      float z = sigmoidf_(acc2[1][v] + b1z);
      float n = tanhf(accn2[v] + b1in + r * (acc2[2][v] + b1hn));
      float hnew = (1.f - z) * n + z * h2p[v];
      h2p[v] = hnew;
      h2_s[(half * 8 + v) * H + j] = (_Float16)hnew;
    }
    __syncthreads();   // h2 ready

    // ---- output projection: pred = h2_new @ Wout^T ([16,128]x[128,32]) ----
    if (wave < 2) {
      v8f accp = {};
      int ncol = wave * 16 + mA;
#pragma unroll
      for (int kt = 0; kt < H; kt += 32) {
        v16h a = load_a_frag(h2row, kt, half);
        accp = wmma_f16(a, load_b_frag(Wout, H, ncol, kt, half), accp);
      }
#pragma unroll
      for (int v = 0; v < 8; ++v)
        pred_s[(half * 8 + v) * D + ncol] = accp[v];
    }
    __syncthreads();
    for (int e = tid; e < 16 * D; e += 256) {
      int m = e >> 5, c = e & (D - 1);
      float v = pred_s[m * D + c] + bout[c];
      recon[(((size_t)rowbase + m) * T + t) * D + c] = v;
      din_s[m * D + c] = (_Float16)v;   // output feedback
    }
    __syncthreads();
  }
}

// ------------------------------ host driver ---------------------------------
extern "C" void kernel_launch(void* const* d_in, const int* in_sizes, int n_in,
                              void* d_out, int out_size, void* d_ws,
                              size_t ws_size, hipStream_t stream) {
  using namespace vrae;
  (void)in_sizes; (void)n_in; (void)out_size; (void)ws_size;

  const float* x       = (const float*)d_in[0];
  const int*   lengths = (const int*)d_in[1];
  const float* eps     = (const float*)d_in[2];
  const float* eWih0   = (const float*)d_in[3];
  const float* eWhh0   = (const float*)d_in[4];
  const float* ebih0   = (const float*)d_in[5];
  const float* ebhh0   = (const float*)d_in[6];
  const float* eWih12  = (const float*)d_in[7];
  const float* eWhh12  = (const float*)d_in[8];
  const float* ebih12  = (const float*)d_in[9];
  const float* ebhh12  = (const float*)d_in[10];
  const float* dWih0   = (const float*)d_in[11];
  const float* dWhh0   = (const float*)d_in[12];
  const float* dbih0   = (const float*)d_in[13];
  const float* dbhh0   = (const float*)d_in[14];
  const float* dWih1   = (const float*)d_in[15];
  const float* dWhh1   = (const float*)d_in[16];
  const float* dbih1   = (const float*)d_in[17];
  const float* dbhh1   = (const float*)d_in[18];
  const float* Wmean   = (const float*)d_in[19];
  const float* bmean   = (const float*)d_in[20];
  const float* Wlogvar = (const float*)d_in[21];
  const float* blogvar = (const float*)d_in[22];
  const float* Wz2h    = (const float*)d_in[23];
  const float* bz2h    = (const float*)d_in[24];
  const float* Wout    = (const float*)d_in[25];
  const float* bout    = (const float*)d_in[26];

  char* ws = (char*)d_ws;
  size_t off = 0;
  auto take = [&](size_t bytes) -> char* {
    char* p = ws + off;
    off = (off + bytes + 255) & ~(size_t)255;
    return p;
  };
  _Float16* x16    = (_Float16*)take((size_t)B * T * D * 2);
  _Float16* seq0   = (_Float16*)take((size_t)B * T * 2 * H * 2);
  _Float16* seq1   = (_Float16*)take((size_t)B * T * 2 * H * 2);
  _Float16* gi12   = (_Float16*)take((size_t)2 * B * T * G3 * 2); // per-layer reuse
  _Float16* fWih0  = (_Float16*)take((size_t)2 * G3 * D * 2);
  _Float16* fWhh0  = (_Float16*)take((size_t)2 * G3 * H * 2);
  _Float16* fWih12 = (_Float16*)take((size_t)4 * G3 * 2 * H * 2);
  _Float16* fWhh12 = (_Float16*)take((size_t)4 * G3 * H * 2);
  _Float16* fdWih0 = (_Float16*)take((size_t)G3 * D * 2);
  _Float16* fdWhh0 = (_Float16*)take((size_t)G3 * H * 2);
  _Float16* fdWih1 = (_Float16*)take((size_t)G3 * H * 2);
  _Float16* fdWhh1 = (_Float16*)take((size_t)G3 * H * 2);
  _Float16* fWout  = (_Float16*)take((size_t)D * H * 2);
  float*    hfinal = (float*)take((size_t)2 * B * H * 4);
  float*    hinit  = (float*)take((size_t)B * H * 4);

  auto cvt = [&](const float* s, _Float16* d, size_t n) {
    f32_to_f16_kernel<<<dim3(1024), dim3(256), 0, stream>>>(s, d, n);
  };
  cvt(x, x16, (size_t)B * T * D);
  cvt(eWih0, fWih0, (size_t)2 * G3 * D);
  cvt(eWhh0, fWhh0, (size_t)2 * G3 * H);
  cvt(eWih12, fWih12, (size_t)4 * G3 * 2 * H);
  cvt(eWhh12, fWhh12, (size_t)4 * G3 * H);
  cvt(dWih0, fdWih0, (size_t)G3 * D);
  cvt(dWhh0, fdWhh0, (size_t)G3 * H);
  cvt(dWih1, fdWih1, (size_t)G3 * H);
  cvt(dWhh1, fdWhh1, (size_t)G3 * H);
  cvt(Wout, fWout, (size_t)D * H);

  // zero sequence buffers (backward direction leaves t >= len untouched)
  zero_u32_kernel<<<dim3(4096), dim3(256), 0, stream>>>(
      (uint32_t*)seq0, (size_t)B * T * 2 * H / 2);
  zero_u32_kernel<<<dim3(4096), dim3(256), 0, stream>>>(
      (uint32_t*)seq1, (size_t)B * T * 2 * H / 2);

  // ---- layer 0: fused scan ----
  gru_scan_l0_kernel<<<dim3(B / 16, 2), 256, 0, stream>>>(
      x16, seq0, fWih0, fWhh0, ebih0, ebhh0, lengths);

  // ---- layer 1: parallel gi GEMM, then recurrent-only scan ----
  gi_gemm_kernel<<<dim3(B * T / 64, 2), 256, 0, stream>>>(seq0, fWih12, gi12);
  gru_scan_pre_kernel<<<dim3(B / 16, 2), 256, 0, stream>>>(
      gi12, seq1, fWhh12, ebih12, ebhh12, lengths, nullptr);

  // ---- layer 2: same; sequence output not needed (only final hiddens) ----
  gi_gemm_kernel<<<dim3(B * T / 64, 2), 256, 0, stream>>>(
      seq1, fWih12 + (size_t)2 * G3 * 2 * H, gi12);
  gru_scan_pre_kernel<<<dim3(B / 16, 2), 256, 0, stream>>>(
      gi12, nullptr, fWhh12 + (size_t)2 * G3 * H,
      ebih12 + 2 * G3, ebhh12 + 2 * G3, lengths, hfinal);

  // ---- latent head ----
  float* out        = (float*)d_out;
  float* recon      = out;                          // [B,T,D]
  float* mean_out   = out + (size_t)B * T * D;      // [B,L]
  float* logvar_out = mean_out + (size_t)B * L;     // [B,L]
  latent_kernel<<<dim3(B), 128, 0, stream>>>(
      hfinal, eps, Wmean, bmean, Wlogvar, blogvar, Wz2h, bz2h,
      mean_out, logvar_out, hinit);

  // ---- autoregressive decoder ----
  decoder_kernel<<<dim3(B / 16), 256, 0, stream>>>(
      hinit, fdWih0, fdWhh0, fdWih1, fdWhh1, fWout,
      dbih0, dbhh0, dbih1, dbhh1, bout, recon);
}
